// MMDLoss_56573309223654
// MI455X (gfx1250) — compile-verified, matrix operand order checked by reading, not run on hardware
//
#include <hip/hip_runtime.h>
#include <hip/hip_bf16.h>

typedef __attribute__((ext_vector_type(8)))  float         v8f;
typedef __attribute__((ext_vector_type(16))) __bf16        v16bf;
typedef __attribute__((ext_vector_type(4)))  unsigned int  u32x4;
typedef __attribute__((ext_vector_type(4)))  int           i32x4;
typedef __attribute__((ext_vector_type(8)))  int           i32x8;

union FragU { u32x4 u[2]; v16bf v; };

// ---- constants for this problem size ----
#define NROWS 4096
#define DDIM  512
#define TM    128      // workgroup M tile
#define TN    64       // workgroup N tile
#define KSTEP 32       // K slab (bf16 elems) per WMMA step
#define NKT   (DDIM / KSTEP)   // 16
#define LDA   40       // padded LDS row stride in bf16 elems (80B: 16B-aligned, conflict-free)
#define GRIDX (NROWS / TN)     // 64
#define GRIDY (NROWS / TM)     // 32
#define NPART (GRIDX * GRIDY)  // 2048

#if defined(__has_builtin)
#if __has_builtin(__builtin_amdgcn_tensor_load_to_lds) && __has_builtin(__builtin_amdgcn_s_wait_tensorcnt)
#define USE_TDM 1
#endif
#endif

// fp32 -> bf16 round-to-nearest-even, packed pair
__device__ __forceinline__ unsigned pk_bf16(float a, float b) {
    unsigned ua = __float_as_uint(a);
    unsigned ub = __float_as_uint(b);
    ua += 0x7fffu + ((ua >> 16) & 1u);
    ub += 0x7fffu + ((ub >> 16) & 1u);
    return ((ua >> 16) & 0xffffu) | (ub & 0xffff0000u);
}

// ---------------- one-shot fp32 -> bf16 conversion (amortized, L2-resident) ----------------
__global__ __launch_bounds__(256) void mmd_cvt(const float* __restrict__ X,
                                               unsigned short* __restrict__ Y) {
    const size_t base = ((size_t)blockIdx.x * 256 + threadIdx.x) * 8;
    float4 f0 = *(const float4*)(X + base);
    float4 f1 = *(const float4*)(X + base + 4);
    u32x4 p;
    p[0] = pk_bf16(f0.x, f0.y);
    p[1] = pk_bf16(f0.z, f0.w);
    p[2] = pk_bf16(f1.x, f1.y);
    p[3] = pk_bf16(f1.z, f1.w);
    *(u32x4*)(Y + base) = p;
}

// ---------------- stats kernels (exact fp32) ----------------
__global__ __launch_bounds__(256) void mmd_rowsq(const float* __restrict__ S,
                                                 const float* __restrict__ T,
                                                 float* __restrict__ rowsq) {
    __shared__ float red[256];
    const int row = blockIdx.x;
    const int t   = threadIdx.x;
    const float* p = (row < NROWS) ? (S + (size_t)row * DDIM)
                                   : (T + (size_t)(row - NROWS) * DDIM);
    float x0 = p[t], x1 = p[t + 256];
    red[t] = x0 * x0 + x1 * x1;
    __syncthreads();
    for (int off = 128; off > 0; off >>= 1) {
        if (t < off) red[t] += red[t + off];
        __syncthreads();
    }
    if (t == 0) rowsq[row] = red[0];
}

__global__ __launch_bounds__(256) void mmd_colsum(const float* __restrict__ S,
                                                  const float* __restrict__ T,
                                                  float* __restrict__ col) {
    __shared__ float red[256];
    const int d = blockIdx.x;
    const int t = threadIdx.x;
    float s = 0.f;
    for (int i = t; i < NROWS; i += 256) s += S[(size_t)i * DDIM + d];
    for (int i = t; i < NROWS; i += 256) s += T[(size_t)i * DDIM + d];
    red[t] = s;
    __syncthreads();
    for (int off = 128; off > 0; off >>= 1) {
        if (t < off) red[t] += red[t + off];
        __syncthreads();
    }
    if (t == 0) col[d] = red[0];
}

// closed-form bandwidth: bw = sum||x||^2 / P - ||mean||^2 ; c = -1/(2 bw^2)
__global__ __launch_bounds__(256) void mmd_bw(const float* __restrict__ rowsq,
                                              const float* __restrict__ col,
                                              float* __restrict__ cexp) {
    __shared__ float red[256];
    const int t = threadIdx.x;
    float s = 0.f;
    for (int i = t; i < 2 * NROWS; i += 256) s += rowsq[i];
    red[t] = s;
    __syncthreads();
    for (int off = 128; off > 0; off >>= 1) {
        if (t < off) red[t] += red[t + off];
        __syncthreads();
    }
    const float sumsq = red[0];
    __syncthreads();
    float m = 0.f;
    for (int i = t; i < DDIM; i += 256) {
        float mu = col[i] * (1.0f / 8192.0f);
        m += mu * mu;
    }
    red[t] = m;
    __syncthreads();
    for (int off = 128; off > 0; off >>= 1) {
        if (t < off) red[t] += red[t + off];
        __syncthreads();
    }
    if (t == 0) {
        float bw = sumsq * (1.0f / 8192.0f) - red[0];
        cexp[0] = -1.0f / (2.0f * bw * bw);
    }
}

#ifdef USE_TDM
// Issue a TDM load of a [tile_rows x 32] bf16 tile (row-major, tensor row stride 512
// elems) into LDS with 4-DWORD padding after every 16 DWORDs (-> 80B LDS row stride).
__device__ __forceinline__ void tdm_load_tile(unsigned lds_addr,
                                              const unsigned short* gptr,
                                              int tile_rows) {
    unsigned long long ga = (unsigned long long)(uintptr_t)gptr;
    u32x4 g0;
    g0[0] = 1u;                                   // count=1, is_restore=0, gather off
    g0[1] = lds_addr;                             // LDS byte address
    g0[2] = (unsigned)(ga & 0xffffffffu);         // global_addr[31:0]
    g0[3] = (unsigned)((ga >> 32) & 0x01ffffffu)  // global_addr[56:32]
            | 0x80000000u;                        // type=2 ("image")
    const unsigned dim0 = 1u << 20, dim1 = 1u << 20;  // huge -> no OOB clipping
    i32x8 g1;
    g1[0] = (int)((1u << 16)      // data_size = 2 bytes
                | (1u << 20)      // pad_enable
                | (3u << 22)      // pad_interval: 16 DWORDs (64B)
                | (3u << 25));    // pad_amount:   4 DWORDs (16B)
    g1[1] = (int)((dim0 & 0xffffu) << 16);                  // tensor_dim0[15:0]
    g1[2] = (int)((dim0 >> 16) | ((dim1 & 0xffffu) << 16)); // dim0[31:16] | dim1[15:0]
    g1[3] = (int)((dim1 >> 16) | (32u << 16));              // dim1[31:16] | tile_dim0=32
    g1[4] = tile_rows;                                      // tile_dim1 (tile_dim2=0)
    g1[5] = DDIM;                                           // tensor_dim0_stride = 512
    g1[6] = 0;
    g1[7] = 0;
    i32x4 z4 = {0, 0, 0, 0};                 // D# groups 2/3 unused for 2-D tiles
    i32x8 z8 = {0, 0, 0, 0, 0, 0, 0, 0};
    __builtin_amdgcn_tensor_load_to_lds(g0, g1, z4, z4, z8, 0);
}
#endif

// ---------------- main fused GEMM + exp + reduce ----------------
__global__ __launch_bounds__(256) void mmd_gemm(const unsigned short* __restrict__ Sb,
                                                const unsigned short* __restrict__ Tb,
                                                const float* __restrict__ a2,
                                                const float* __restrict__ b2,
                                                const float* __restrict__ cexp,
                                                float* __restrict__ partials) {
    __shared__ unsigned short smA[2][TM * LDA];  // 2 x 10240 B (128 x 32 bf16, padded)
    __shared__ unsigned short smB[2][TN * LDA];  // 2 x  5120 B ( 64 x 32 bf16, padded)
    __shared__ float smRed[8];

    const int tid   = threadIdx.x;
    const int mBase = blockIdx.y * TM;
    const int nBase = blockIdx.x * TN;
    const float c   = cexp[0];

    const int lane = tid & 31;
    const int wid  = tid >> 5;
    const int wm   = wid & 3;   // 4 waves over M -> 32 rows each
    const int wn   = wid >> 2;  // 2 waves over N -> 32 cols each
    const int lr   = lane & 15;
    const int lh   = lane >> 4;

#ifdef USE_TDM
    auto stage = [&](int buf, int kt) {
        if (wid == 0) {
            const int kOff = kt * KSTEP;
            tdm_load_tile((unsigned)(uintptr_t)&smA[buf][0],
                          Sb + (size_t)mBase * DDIM + kOff, TM);
            tdm_load_tile((unsigned)(uintptr_t)&smB[buf][0],
                          Tb + (size_t)nBase * DDIM + kOff, TN);
        }
    };
    auto stage_wait = [&]() {
        if (wid == 0) __builtin_amdgcn_s_wait_tensorcnt(0);
    };
#else
    // fallback: pure b128 copy (no conversion ALU). 768 16B-granules per tile pair,
    // 3 per thread: granule g -> row g>>2, quarter-row q = g&3.
    auto stage = [&](int buf, int kt) {
        const int kOff = kt * KSTEP;
        for (int it = 0; it < 2; ++it) {   // A: 512 granules
            int g = it * 256 + tid, r = g >> 2, q = g & 3;
            u32x4 v = *(const u32x4*)(Sb + (size_t)(mBase + r) * DDIM + kOff + q * 8);
            *(u32x4*)(&smA[buf][r * LDA + q * 8]) = v;
        }
        {                                   // B: 256 granules
            int g = tid, r = g >> 2, q = g & 3;
            u32x4 v = *(const u32x4*)(Tb + (size_t)(nBase + r) * DDIM + kOff + q * 8);
            *(u32x4*)(&smB[buf][r * LDA + q * 8]) = v;
        }
    };
    auto stage_wait = [&]() {};
#endif

    v8f acc[2][2] = {};

    stage(0, 0);
    stage_wait();
    __syncthreads();

    for (int kt = 0; kt < NKT; ++kt) {
        const int cur = kt & 1, nxt = cur ^ 1;
        if (kt + 1 < NKT) stage(nxt, kt + 1);   // overlaps with compute below

        FragU aF[2], bF[2];
        for (int s = 0; s < 2; ++s) {
            const unsigned short* pa = &smA[cur][(wm * 32 + s * 16 + lr) * LDA + lh * 8];
            aF[s].u[0] = *(const u32x4*)pa;
            aF[s].u[1] = *(const u32x4*)(pa + 16);
            const unsigned short* pb = &smB[cur][(wn * 32 + s * 16 + lr) * LDA + lh * 8];
            bF[s].u[0] = *(const u32x4*)pb;
            bF[s].u[1] = *(const u32x4*)(pb + 16);
        }
        for (int i = 0; i < 2; ++i)
            for (int j = 0; j < 2; ++j)
                acc[i][j] = __builtin_amdgcn_wmma_f32_16x16x32_bf16(
                    false, aF[i].v, false, bF[j].v, (short)0, acc[i][j], false, false);

        if (kt + 1 < NKT) stage_wait();
        __syncthreads();
    }

    // epilogue: d2 = a2 + b2 - 2ab, clamp, exp, sum
    float b2v[2];
    for (int j = 0; j < 2; ++j) b2v[j] = b2[nBase + wn * 32 + j * 16 + lr];

    float wsum = 0.f;
    for (int i = 0; i < 2; ++i) {
        const int rbase = mBase + wm * 32 + i * 16 + 8 * lh;
        for (int e = 0; e < 8; ++e) {
            const float av = a2[rbase + e];
            for (int j = 0; j < 2; ++j) {
                float d2 = av + b2v[j] - 2.0f * acc[i][j][e];
                d2 = fmaxf(d2, 0.0f);
                wsum += __expf(c * d2);
            }
        }
    }
    for (int off = 16; off > 0; off >>= 1) wsum += __shfl_xor(wsum, off, 32);
    if (lane == 0) smRed[wid] = wsum;
    __syncthreads();
    if (tid == 0) {
        float s = 0.f;
        for (int w = 0; w < 8; ++w) s += smRed[w];  // fixed order -> deterministic
        partials[blockIdx.y * gridDim.x + blockIdx.x] = s;
    }
}

// fixed-order final reduction; loss = sum / n^2
__global__ __launch_bounds__(256) void mmd_final(const float* __restrict__ part,
                                                 float* __restrict__ out) {
    __shared__ float red[256];
    const int t = threadIdx.x;
    float s = 0.f;
    for (int i = t; i < NPART; i += 256) s += part[i];
    red[t] = s;
    __syncthreads();
    for (int off = 128; off > 0; off >>= 1) {
        if (t < off) red[t] += red[t + off];
        __syncthreads();
    }
    if (t == 0) out[0] = red[0] * (1.0f / (4096.0f * 4096.0f));
}

extern "C" void kernel_launch(void* const* d_in, const int* in_sizes, int n_in,
                              void* d_out, int out_size, void* d_ws, size_t ws_size,
                              hipStream_t stream) {
    (void)in_sizes; (void)n_in; (void)out_size; (void)ws_size;
    const float* S = (const float*)d_in[0];
    const float* T = (const float*)d_in[1];

    // ws layout: [Sb bf16 4MB][Tb bf16 4MB][stats floats ~44KB]
    unsigned short* Sb = (unsigned short*)d_ws;
    unsigned short* Tb = Sb + (size_t)NROWS * DDIM;
    float* stats = (float*)(Tb + (size_t)NROWS * DDIM);
    float* rowsq = stats;          // 8192: ||x||^2 concat rows (a2 | b2)
    float* col   = stats + 8192;   // 512 column sums
    float* cexp  = stats + 8704;   // scalar -1/(2 bw^2)
    float* part  = stats + 8720;   // 2048 per-workgroup partials

    const int cvtBlocks = (NROWS * DDIM) / (256 * 8);  // 1024
    mmd_cvt   <<<cvtBlocks, 256, 0, stream>>>(S, Sb);
    mmd_cvt   <<<cvtBlocks, 256, 0, stream>>>(T, Tb);
    mmd_rowsq <<<2 * NROWS, 256, 0, stream>>>(S, T, rowsq);
    mmd_colsum<<<DDIM,      256, 0, stream>>>(S, T, col);
    mmd_bw    <<<1,         256, 0, stream>>>(rowsq, col, cexp);
    mmd_gemm  <<<dim3(GRIDX, GRIDY), 256, 0, stream>>>(Sb, Tb, rowsq, rowsq + NROWS,
                                                       cexp, part);
    mmd_final <<<1,         256, 0, stream>>>(part, (float*)d_out);
}
